// MultiHeadSelfAttention_62354335204093
// MI455X (gfx1250) — matrix operand order repaired and checked by reading, not compiled, not run on hardware
//
#include <hip/hip_runtime.h>

#define B_  4
#define L_  2048
#define H_  1024
#define NH_ 16
#define DK_ 64
#define S_  (B_ * L_)   // 8192 flattened rows

typedef __attribute__((ext_vector_type(16))) _Float16 v16h;
typedef __attribute__((ext_vector_type(8)))  float    v8f;
typedef __attribute__((ext_vector_type(4)))  unsigned int u32x4;
typedef __attribute__((ext_vector_type(4)))  int      i32x4;
typedef __attribute__((ext_vector_type(8)))  int      i32x8;

#if __has_builtin(__builtin_amdgcn_tensor_load_to_lds)
#define USE_TDM 1
#else
#define USE_TDM 0
#endif

// Wave-local LDS store->load ordering fence.
#if __has_builtin(__builtin_amdgcn_fence)
#define WAVE_LDS_FENCE() __builtin_amdgcn_fence(__ATOMIC_ACQ_REL, "wavefront")
#else
#define WAVE_LDS_FENCE() __syncthreads()
#endif

__device__ __forceinline__ int lane_id() { return (int)(threadIdx.x & 31u); }

__device__ __forceinline__ v8f wmma16(v16h a, v16h b, v8f c) {
  // D = A(16x32 f16) * B(32x16 f16) + C(16x16 f32)
  return __builtin_amdgcn_wmma_f32_16x16x32_f16(false, a, false, b, (short)0, c,
                                                false, false);
}

// A fragment: A[m][k] with m = lane%16; K split per CDNA5 ISA layout:
// VGPR j<4 -> K = 2j + 8*(lane/16), VGPR j>=4 -> K = 16 + 2(j-4) + 8*(lane/16)
// Contiguous 8-half runs per lane -> 2x ds_load_b128 when rows are 16B aligned.
__device__ __forceinline__ v16h load_a_frag(const _Float16* base, int ld) {
  const int l = lane_id(), m = l & 15, hf = l >> 4;
  v16h a;
#pragma unroll
  for (int j = 0; j < 8; ++j) {
    const int kb = (j < 4 ? 2 * j : 16 + 2 * (j - 4)) + 8 * hf;
    a[2 * j]     = base[m * ld + kb];
    a[2 * j + 1] = base[m * ld + kb + 1];
  }
  return a;
}

// B fragment, source K-major (row k, col n): B[k][n] = base[k*ld + n].
// Per lane: column n = lane%16, K in the same split order as the A fragment.
__device__ __forceinline__ v16h load_b_frag_kn(const _Float16* base, int ld) {
  const int l = lane_id(), n = l & 15, hf = l >> 4;
  v16h b;
#pragma unroll
  for (int j = 0; j < 8; ++j) {
    const int kb = (j < 4 ? 2 * j : 16 + 2 * (j - 4)) + 8 * hf;
    b[2 * j]     = base[kb * ld + n];
    b[2 * j + 1] = base[(kb + 1) * ld + n];
  }
  return b;
}

#if USE_TDM
// Tensor Data Mover: DMA a 64-row x 64-col f16 tile (row stride DK_ elements)
// from global into LDS, with TDM padding of 4 DWORDs after every 32 DWORDs
// (one row) -> LDS row stride 72 halfs (144B, 16B aligned).
// D# bitfields per CDNA5 ISA ch.8 (group0: flags/lds/global/type, group1:
// data_size=2B, pad cfg, tensor/tile dims + stride).
__device__ __forceinline__ void tdm_load_v_tile(const _Float16* gsrc,
                                                void* lds_dst) {
  const unsigned long long ga = (unsigned long long)(size_t)gsrc;
  u32x4 g0;
  g0[0] = 1u;                                  // count = 1 valid descriptor
  g0[1] = (unsigned int)(size_t)lds_dst;       // lds_addr (bytes)
  g0[2] = (unsigned int)(ga & 0xFFFFFFFFu);    // global_addr[31:0]
  g0[3] = (unsigned int)((ga >> 32) & 0x01FFFFFFu) | (2u << 30);  // type=2
  union { i32x8 v; unsigned long long q[4]; } g1;
  g1.q[0] = (1ULL << 16)                       // data_size = 2 bytes
          | (1ULL << 20)                       // pad_enable
          | (4ULL << 22)                       // pad_interval: 32 DWORDs
          | (3ULL << 25)                       // pad_amount: 4 DWORDs
          | ((unsigned long long)DK_ << 48);   // tensor_dim0[15:0]
  g1.q[1] = ((unsigned long long)L_ << 16)     // tensor_dim1
          | ((unsigned long long)DK_ << 48);   // tile_dim0 = 64
  g1.q[2] = 64ULL                              // tile_dim1 = 64 rows
          | ((unsigned long long)DK_ << 32);   // tensor_dim0_stride = 64
  g1.q[3] = 0ULL;
  const i32x4 gz = {};
#if defined(__clang_major__) && (__clang_major__ >= 23)
  const i32x8 gz8 = {};
  __builtin_amdgcn_tensor_load_to_lds(g0, g1.v, gz, gz, gz8, 0);
#else
  __builtin_amdgcn_tensor_load_to_lds(g0, g1.v, gz, gz, 0);
#endif
}
#endif

// ---------------------------------------------------------------------------
// Kernel 0: zero the atomic accumulators (weights region of d_out, cmean)
// ---------------------------------------------------------------------------
__global__ __launch_bounds__(256) void zero_kernel(float* __restrict__ wacc,
                                                   float* __restrict__ cmean) {
  const int i = blockIdx.x * 256 + threadIdx.x;
  if (i < B_ * L_) wacc[i] = 0.0f;
  if (i < B_ * H_) cmean[i] = 0.0f;
}

// ---------------------------------------------------------------------------
// Kernel 1: fused QKV projection GEMM.  Y[l,o] = sum_h x[l,h] * W[o,h] + b[o]
// Block: 256 threads (8 waves). Tile: 64 rows x 128 cols (4 WMMAs per wave
// per 32-deep k-step -> 4x W-panel reuse). W tile stored TRANSPOSED in LDS
// so B fragments read with ds_load_b128.
// Output written as (B, NH, L, DK) f16 for the attention kernel.
// ---------------------------------------------------------------------------
#define XT_LD 40    // 80B row stride: 16B aligned, conflict-free A reads
#define WT_LD 136   // 272B row stride: 16B aligned, <=2-way conflicts

__global__ __launch_bounds__(256) void qkv_kernel(
    const float* __restrict__ x,
    const float* __restrict__ Wq, const float* __restrict__ Wk,
    const float* __restrict__ Wv,
    const float* __restrict__ bq, const float* __restrict__ bk,
    const float* __restrict__ bv,
    _Float16* __restrict__ Qo, _Float16* __restrict__ Ko,
    _Float16* __restrict__ Vo) {
  __shared__ _Float16 Xt[64][XT_LD];     // x tile, row-major (m, k)
  __shared__ _Float16 WtT[32][WT_LD];    // W tile, TRANSPOSED (k, o_local)

  const int z = blockIdx.z;
  const float* W    = (z == 0) ? Wq : (z == 1) ? Wk : Wv;
  const float* bias = (z == 0) ? bq : (z == 1) ? bk : bv;
  _Float16*    out  = (z == 0) ? Qo : (z == 1) ? Ko : Vo;

  const int row0 = blockIdx.y * 64;          // flattened (b*L + l) row base
  const int col0 = blockIdx.x * 128;         // output-feature base
  const int wave = threadIdx.x >> 5;
  const int lane = threadIdx.x & 31;

  v8f acc[4] = {};
  for (int k0 = 0; k0 < H_; k0 += 32) {
    __syncthreads();
    for (int i = threadIdx.x; i < 64 * 32; i += 256) {
      const int r = i >> 5, c = i & 31;
      Xt[r][c] = (_Float16)x[(size_t)(row0 + r) * H_ + k0 + c];
    }
    // Read W rows coalesced (consecutive threads -> consecutive h), store
    // transposed into LDS (scatter once; vector-read many times).
    for (int i = threadIdx.x; i < 128 * 32; i += 256) {
      const int r = i >> 5, c = i & 31;   // r: o_local, c: k_local
      WtT[c][r] = (_Float16)W[(size_t)(col0 + r) * H_ + k0 + c];
    }
    __syncthreads();
    const v16h bfr = load_b_frag_kn(&WtT[0][wave * 16], WT_LD);
#pragma unroll
    for (int rt = 0; rt < 4; ++rt) {
      const v16h a = load_a_frag(&Xt[rt * 16][0], XT_LD);
      acc[rt] = wmma16(a, bfr, acc[rt]);
    }
  }

  // Scatter D fragments into (B, NH, L, DK) layout with bias add.
  const int n = lane & 15, hf = lane >> 4;
  const int o = col0 + wave * 16 + n;
  const int head = o >> 6, d = o & 63;
  const float bia = bias[o];
#pragma unroll
  for (int rt = 0; rt < 4; ++rt) {
#pragma unroll
    for (int r = 0; r < 8; ++r) {
      const int srow = row0 + rt * 16 + r + 8 * hf;
      const int bb = srow >> 11;       // / L_
      const int l  = srow & (L_ - 1);
      out[(((size_t)(bb * NH_ + head) * L_ + l) * DK_) + d] =
          (_Float16)(acc[rt][r] + bia);
    }
  }
}

// ---------------------------------------------------------------------------
// Kernel 2: attention. All 8 waves of a block share one (b, head); each wave
// owns one 16-query tile. 64-wide k-tiles: K staged transposed in LDS once
// per block; V tiles DMA'd by the Tensor Data Mover.
// TWO passes over K (online softmax):
//   pass 1: per-lane running (m, z) with exp2-rescaling; cross-lane (m,z)
//           merge at the end.
//   pass 2: probs -> weights colsums (atomics), P*V WMMA, cmean colsums.
// ---------------------------------------------------------------------------
#define KT_LD 72    // KtT row stride (d-major): 144B, 16B aligned
#define VT_LD 72    // Vt row stride (matches TDM pad): 144B, 16B aligned
#define PT_LD 72

__global__ __launch_bounds__(256) void attn_kernel(
    const _Float16* __restrict__ Qb, const _Float16* __restrict__ Kb,
    const _Float16* __restrict__ Vb, float* __restrict__ wacc,
    float* __restrict__ cmean) {
  __shared__ _Float16 KtT[64][KT_LD];      // K tile transposed: [d][kcol 0..63]
  __shared__ _Float16 Vt[64][VT_LD];       // V tile: [krow][d]
  __shared__ _Float16 Pt[8][16][PT_LD];    // per-wave P transpose tile (16x64)

  const int t = (int)threadIdx.x;
  const int wave = t >> 5;
  const int lane = t & 31;
  const int hf = lane >> 4, ln = lane & 15;
  const int gw = blockIdx.x * 8 + wave;
  const int bh = gw >> 7;              // block-uniform: 8 waves share (b,head)
  const int qt = gw & 127;
  const int b = bh >> 4, head = bh & 15;
  const int q0 = qt * 16;
  const bool wave0 = ((__builtin_amdgcn_readfirstlane(t) >> 5) == 0);

  const _Float16* Qh = Qb + (size_t)bh * L_ * DK_;
  const _Float16* Kh = Kb + (size_t)bh * L_ * DK_;
  const _Float16* Vh = Vb + (size_t)bh * L_ * DK_;
  const float kSc = 0.125f * 1.4426950408889634f;  // 1/sqrt(64) * log2(e)

  const v16h aq0 = load_a_frag(Qh + (size_t)q0 * DK_, DK_);
  const v16h aq1 = load_a_frag(Qh + (size_t)q0 * DK_ + 32, DK_);

  float m8[8], z8[8];
#pragma unroll
  for (int r = 0; r < 8; ++r) { m8[r] = -1e30f; z8[r] = 0.0f; }

  // ---- pass 1: online (m, z), per lane over its column subset ----
  for (int k0 = 0; k0 < L_; k0 += 64) {
    __syncthreads();
#pragma unroll
    for (int j = 0; j < 4; ++j) {    // stage 64x64 K tile, transposed
      const int c = t + j * 256;
      const int row = c >> 3, db = (c & 7) * 8;
      union { uint4 u; _Float16 h[8]; } cv;
      cv.u = ((const uint4*)(Kh + (size_t)k0 * DK_))[c];
#pragma unroll
      for (int e = 0; e < 8; ++e) KtT[db + e][row] = cv.h[e];
    }
    __syncthreads();
    v8f s[4];
#pragma unroll
    for (int sub = 0; sub < 4; ++sub) {
      const v8f zf = {};
      s[sub] = wmma16(aq0, load_b_frag_kn(&KtT[0][sub * 16],  KT_LD), zf);
      s[sub] = wmma16(aq1, load_b_frag_kn(&KtT[32][sub * 16], KT_LD), s[sub]);
    }
#pragma unroll
    for (int r = 0; r < 8; ++r) {
      const float v0 = s[0][r] * kSc, v1 = s[1][r] * kSc;
      const float v2 = s[2][r] * kSc, v3 = s[3][r] * kSc;
      const float mn =
          fmaxf(m8[r], fmaxf(fmaxf(v0, v1), fmaxf(v2, v3)));
      z8[r] = z8[r] * __builtin_amdgcn_exp2f(m8[r] - mn) +
              __builtin_amdgcn_exp2f(v0 - mn) +
              __builtin_amdgcn_exp2f(v1 - mn) +
              __builtin_amdgcn_exp2f(v2 - mn) +
              __builtin_amdgcn_exp2f(v3 - mn);
      m8[r] = mn;
    }
    if (k0 + 64 < L_)
      __builtin_prefetch(Kh + (size_t)(k0 + 64) * DK_, 0, 0);
  }
  // Cross-lane (m, z) merge within each 16-lane half.
#pragma unroll
  for (int r = 0; r < 8; ++r) {
#pragma unroll
    for (int off = 8; off >= 1; off >>= 1) {
      const float mo = __shfl_xor(m8[r], off);
      const float zo = __shfl_xor(z8[r], off);
      const float mn = fmaxf(m8[r], mo);
      z8[r] = z8[r] * __builtin_amdgcn_exp2f(m8[r] - mn) +
              zo    * __builtin_amdgcn_exp2f(mo    - mn);
      m8[r] = mn;
    }
    z8[r] = 1.0f / z8[r];
  }

  // ---- pass 2: probs -> weights colsums, P*V, cmean ----
  v8f oacc[4] = {};
  const float wScale = 1.0f / (float)(NH_ * L_);
  for (int k0 = 0; k0 < L_; k0 += 64) {
    __syncthreads();
#if USE_TDM
    if (wave0)  // uniform scalar branch; TDM ignores EXEC, issue once
      tdm_load_v_tile(Vh + (size_t)k0 * DK_, &Vt[0][0]);
#endif
#pragma unroll
    for (int j = 0; j < 4; ++j) {
      const int c = t + j * 256;
      const int row = c >> 3, db = (c & 7) * 8;
      union { uint4 u; _Float16 h[8]; } cv;
      cv.u = ((const uint4*)(Kh + (size_t)k0 * DK_))[c];
#pragma unroll
      for (int e = 0; e < 8; ++e) KtT[db + e][row] = cv.h[e];
    }
#if !USE_TDM
#pragma unroll
    for (int j = 0; j < 4; ++j) {
      const int c = t + j * 256;
      const int row = c >> 3, db = (c & 7) * 8;
      union { uint4 u; _Float16 h[8]; } cv;
      cv.u = ((const uint4*)(Vh + (size_t)k0 * DK_))[c];
      *(uint4*)&Vt[row][db] = cv.u;
    }
#endif
#if USE_TDM
    if (wave0) __builtin_amdgcn_s_wait_tensorcnt(0);
#endif
    __syncthreads();

    v8f s[4];
#pragma unroll
    for (int sub = 0; sub < 4; ++sub) {
      const v8f zf = {};
      s[sub] = wmma16(aq0, load_b_frag_kn(&KtT[0][sub * 16],  KT_LD), zf);
      s[sub] = wmma16(aq1, load_b_frag_kn(&KtT[32][sub * 16], KT_LD), s[sub]);
    }
#pragma unroll
    for (int sub = 0; sub < 4; ++sub) {
      float p[8];
      float cs = 0.0f;
#pragma unroll
      for (int r = 0; r < 8; ++r) {
        p[r] = __builtin_amdgcn_exp2f(s[sub][r] * kSc - m8[r]) * z8[r];
        cs += p[r];
      }
      cs += __shfl_xor(cs, 16);   // combine lane halves (rows 0-7 + 8-15)
      if (lane < 16)
        atomicAdd(&wacc[b * L_ + k0 + sub * 16 + ln], cs * wScale);
#pragma unroll
      for (int r = 0; r < 8; ++r)
        Pt[wave][r + 8 * hf][sub * 16 + ln] = (_Float16)p[r];
    }
    WAVE_LDS_FENCE();  // wave-local LDS RAW ordering (store -> transposed load)
    const v16h ap0 = load_a_frag(&Pt[wave][0][0],  PT_LD);
    const v16h ap1 = load_a_frag(&Pt[wave][0][32], PT_LD);
#pragma unroll
    for (int tt = 0; tt < 4; ++tt) {
      oacc[tt] = wmma16(ap0, load_b_frag_kn(&Vt[0][tt * 16],  VT_LD), oacc[tt]);
      oacc[tt] = wmma16(ap1, load_b_frag_kn(&Vt[32][tt * 16], VT_LD), oacc[tt]);
    }

    if (k0 + 64 < L_)
      __builtin_prefetch(Kh + (size_t)(k0 + 64) * DK_, 0, 0);
  }

  // context column means -> cmean[b, head*64 + d]
  const float lScale = 1.0f / (float)L_;
#pragma unroll
  for (int tt = 0; tt < 4; ++tt) {
    float c = 0.0f;
#pragma unroll
    for (int r = 0; r < 8; ++r) c += oacc[tt][r];
    c += __shfl_xor(c, 16);
    if (lane < 16)
      atomicAdd(&cmean[b * H_ + head * 64 + tt * 16 + ln], c * lScale);
  }
}

// ---------------------------------------------------------------------------
// Kernel 3: pooled = cmean @ Wo^T + bo.  One block per output element;
// threads stride h -> fully coalesced Wo reads; shfl + LDS reduction.
// (The mean over L is folded through the linear layer, so the full output
// projection GEMM is never needed.)
// ---------------------------------------------------------------------------
__global__ __launch_bounds__(256) void pooled_kernel(
    const float* __restrict__ cmean, const float* __restrict__ Wo,
    const float* __restrict__ bo, float* __restrict__ pooled) {
  __shared__ float red[8];
  const int i = blockIdx.x;            // 0 .. B*H-1
  const int b = i >> 10, o = i & (H_ - 1);
  const int t = (int)threadIdx.x;
  float partial = 0.0f;
  for (int h = t; h < H_; h += 256)
    partial += cmean[b * H_ + h] * Wo[(size_t)o * H_ + h];
#pragma unroll
  for (int off = 16; off >= 1; off >>= 1)
    partial += __shfl_xor(partial, off);
  if ((t & 31) == 0) red[t >> 5] = partial;
  __syncthreads();
  if (t == 0) {
    float sum = bo[o];
#pragma unroll
    for (int w = 0; w < 8; ++w) sum += red[w];
    pooled[b * H_ + o] = sum;
  }
}

extern "C" void kernel_launch(void* const* d_in, const int* in_sizes, int n_in,
                              void* d_out, int out_size, void* d_ws,
                              size_t ws_size, hipStream_t stream) {
  (void)in_sizes; (void)n_in; (void)out_size; (void)ws_size;
  const float* x  = (const float*)d_in[0];
  const float* Wq = (const float*)d_in[1];
  const float* bq = (const float*)d_in[2];
  const float* Wk = (const float*)d_in[3];
  const float* bk = (const float*)d_in[4];
  const float* Wv = (const float*)d_in[5];
  const float* bv = (const float*)d_in[6];
  const float* Wo = (const float*)d_in[7];
  const float* bo = (const float*)d_in[8];
  float* out = (float*)d_out;  // [0,4096): pooled, [4096,12288): weights

  char* ws = (char*)d_ws;
  _Float16* Q = (_Float16*)(ws);
  _Float16* K = (_Float16*)(ws + (size_t)16 * 1024 * 1024);
  _Float16* V = (_Float16*)(ws + (size_t)32 * 1024 * 1024);
  float* cmean = (float*)(ws + (size_t)48 * 1024 * 1024);

  float* wacc = out + B_ * H_;

  zero_kernel<<<(B_ * L_ + 255) / 256, 256, 0, stream>>>(wacc, cmean);

  dim3 gq(H_ / 128, S_ / 64, 3);
  qkv_kernel<<<gq, 256, 0, stream>>>(x, Wq, Wk, Wv, bq, bk, bv, Q, K, V);

  attn_kernel<<<(B_ * NH_ * (L_ / 16)) / 8, 256, 0, stream>>>(Q, K, V, wacc,
                                                              cmean);

  pooled_kernel<<<B_ * H_, 256, 0, stream>>>(cmean, Wo, bo, out);
}
